// AttentionLayer_47854525612133
// MI455X (gfx1250) — matrix operand order repaired and checked by reading, not compiled, not run on hardware
//
#include <hip/hip_runtime.h>
#include <hip/hip_bf16.h>

typedef _Float16 v16h __attribute__((ext_vector_type(16)));
typedef _Float16 h8v  __attribute__((ext_vector_type(8)));
typedef _Float16 h4v  __attribute__((ext_vector_type(4)));
typedef float    v8f  __attribute__((ext_vector_type(8)));
typedef float    f4v  __attribute__((ext_vector_type(4)));

#define B_  64
#define TE_ 512
#define TD_ 512
#define H_  256

// ---------------------------------------------------------------------------
// WMMA helpers (CDNA5: D = A(16x32 f16) * B(32x16 f16) + C(16x16 f32))
// ---------------------------------------------------------------------------
__device__ __forceinline__ v8f wmma_f16(v16h a, v16h b, v8f c) {
    return __builtin_amdgcn_wmma_f32_16x16x32_f16(
        /*neg_a=*/false, a, /*neg_b=*/false, b,
        /*c_mod=*/(short)0, c, /*reuse_a=*/false, /*reuse_b=*/false);
}

// A-fragment (16x32, f16) from row-major storage: lane L holds row (L&15);
// halves 0..7 = k(8*hi .. 8*hi+7), halves 8..15 = k(16+8*hi .. 23+8*hi).
__device__ __forceinline__ v16h load_afrag(const _Float16* base, int stride, int lane) {
    const int hi = lane >> 4;
    const int r  = lane & 15;
    const _Float16* p = base + (size_t)r * stride;
    h8v lo = *(const h8v*)(p + 8 * hi);
    h8v up = *(const h8v*)(p + 16 + 8 * hi);
    return __builtin_shufflevector(lo, up, 0,1,2,3,4,5,6,7,8,9,10,11,12,13,14,15);
}

// B-fragment (32x16, f16) from "column rows" storage (row n = column n of B,
// k contiguous): lane holds column (lane&15); halves 0..7 = k(8*hi .. 8*hi+7), halves 8..15 = k(16+8*hi .. 23+8*hi).
__device__ __forceinline__ v16h load_bfrag(const _Float16* base, int stride, int lane) {
    const int hi = lane >> 4;
    const int n  = lane & 15;
    const _Float16* p = base + (size_t)n * stride;
    h8v lo = *(const h8v*)(p + 8 * hi);
    h8v up = *(const h8v*)(p + 16 + 8 * hi);
    return __builtin_shufflevector(lo, up, 0,1,2,3,4,5,6,7,8,9,10,11,12,13,14,15);
}

// Async global->LDS copy, 16 bytes per lane (GVS addressing: SGPR base +
// per-lane 32-bit offset; LDS dest byte-offset in a VGPR). Tracked on ASYNCcnt.
__device__ __forceinline__ void async_ld_b128(unsigned ldsOff, const void* gbase,
                                              unsigned gOff) {
    asm volatile("global_load_async_to_lds_b128 %0, %1, %2"
                 :
                 : "v"(ldsOff), "v"(gOff), "s"(gbase)
                 : "memory");
}

// Generic LDS pointer -> wave-relative LDS byte offset (flat-aperture rule:
// low 32 bits of a generic LDS address are the LDS offset).
__device__ __forceinline__ unsigned lds_off(const void* p) {
    return (unsigned)(uintptr_t)p;
}

// ---------------------------------------------------------------------------
// Kernel 0: f32 -> f16 conversion of encoder & decoder; copy decoder into
// out[..., H:2H] (the concat half). Pure streaming, b128 loads/stores.
// ---------------------------------------------------------------------------
__global__ void attn_convert_kernel(const float* __restrict__ enc,
                                    const float* __restrict__ dec,
                                    _Float16* __restrict__ ench,
                                    _Float16* __restrict__ dech,
                                    float* __restrict__ out) {
    const size_t n4 = (size_t)B_ * TE_ * H_ / 4;
    size_t i = (size_t)blockIdx.x * blockDim.x + threadIdx.x;
    const size_t stride = (size_t)gridDim.x * blockDim.x;
    for (; i < n4; i += stride) {
        f4v e = ((const f4v*)enc)[i];
        f4v d = ((const f4v*)dec)[i];
        h4v he, hd;
#pragma unroll
        for (int j = 0; j < 4; j++) { he[j] = (_Float16)e[j]; hd[j] = (_Float16)d[j]; }
        ((h4v*)ench)[i] = he;
        ((h4v*)dech)[i] = hd;
        const size_t base = i * 4;
        const size_t bd = base >> 8;       // row index over B*TD
        const int    h  = (int)(base & 255);
        *(f4v*)(out + bd * (2 * H_) + H_ + h) = d;   // decoder concat half
    }
}

// ---------------------------------------------------------------------------
// Kernel 1: proj = encoder @ Wa  (f16 in, f32 accumulate, f16 out)
// Block = 128 flat rows of [B*TE, H]; 8 waves x 16 rows.
// Wa^T staged in LDS (f16) in two 128-column halves (66 KB each).
// ---------------------------------------------------------------------------
__global__ __launch_bounds__(256) void attn_proj_kernel(
    const _Float16* __restrict__ ench,
    const float* __restrict__ Wa,
    _Float16* __restrict__ projh) {
    extern __shared__ _Float16 sm[];   // 128 rows x 264 halfs (WaT half)
    const int tid  = threadIdx.x;
    const int lane = tid & 31;
    const int wid  = tid >> 5;
    const int hi   = lane >> 4;
    const int ln   = lane & 15;
    const size_t row0 = (size_t)blockIdx.x * 128;

    // A-fragments: 16 encoder rows, K = 256
    v16h af[8];
    {
        const _Float16* abase = ench + (row0 + wid * 16) * H_;
#pragma unroll
        for (int kc = 0; kc < 8; kc++) af[kc] = load_afrag(abase + kc * 32, H_, lane);
    }
    const v8f zf = {};

    for (int gh = 0; gh < 2; gh++) {
        __syncthreads();
        // Stage WaT[g - gh*128][h] = (f16)Wa[h][g], coalesced f32 reads.
        for (int i = 0; i < 128; i++) {
            int idx = i * 256 + tid;           // 32768 elements per half
            int h   = idx >> 7;
            int gl  = idx & 127;
            sm[gl * 264 + h] = (_Float16)Wa[h * 256 + gh * 128 + gl];
        }
        __syncthreads();
#pragma unroll
        for (int gt = 0; gt < 8; gt++) {
            v8f acc = zf;
#pragma unroll
            for (int kc = 0; kc < 8; kc++) {
                v16h bf = load_bfrag(sm + gt * 16 * 264 + kc * 32, 264, lane);
                acc = wmma_f16(af[kc], bf, acc);
            }
            const int gcol = gh * 128 + gt * 16 + ln;
#pragma unroll
            for (int r = 0; r < 8; r++)
                projh[(row0 + wid * 16 + r + 8 * hi) * H_ + gcol] = (_Float16)acc[r];
        }
    }
}

// ---------------------------------------------------------------------------
// Kernel 2: fused flash-style attention.
// Block = (b, 128 td rows); 8 waves x 16 rows. Loop over TE in 32-wide tiles:
//   scores = dec @ proj^T (WMMA), running-max exp, P (f16) via LDS relayout,
//   C_acc += P @ enc (WMMA). Epilogue: C = C_acc / rowsum, store f32.
// proj tile staged with GLOBAL_LOAD_ASYNC_TO_LDS_B128 (ASYNCcnt); next tile
// prefetched with global_prefetch while current tile computes.
// ---------------------------------------------------------------------------
__global__ __launch_bounds__(256) void attn_fused_kernel(
    const _Float16* __restrict__ ench,
    const _Float16* __restrict__ dech,
    const _Float16* __restrict__ projh,
    const float* __restrict__ mask,
    float* __restrict__ out) {
    extern __shared__ _Float16 sm[];
    const int PROJ_OFF = 0;                  // 32 x 264 halfs (proj rows, row-major)
    const int ENCT_OFF = 32 * 264;           // 256 x 40 halfs (enc tile transposed)
    const int P_OFF    = ENCT_OFF + 256 * 40;// 8 waves x 16 x 40 halfs (P buffers)

    const int tid  = threadIdx.x;
    const int lane = tid & 31;
    const int wid  = tid >> 5;
    const int hi   = lane >> 4;
    const int ln   = lane & 15;
    const int b    = blockIdx.x >> 2;
    const int td0  = (blockIdx.x & 3) * 128;

    // Decoder A-fragments for this wave's 16 rows (persist across e-loop).
    v16h afdec[8];
    {
        const _Float16* dbase = dech + ((size_t)(b * TD_ + td0 + wid * 16)) * H_;
#pragma unroll
        for (int kc = 0; kc < 8; kc++) afdec[kc] = load_afrag(dbase + kc * 32, H_, lane);
    }

    const v8f zf = {};
    v8f Cacc[16];
#pragma unroll
    for (int g = 0; g < 16; g++) Cacc[g] = zf;
    float Ssum[8], mrow[8];
#pragma unroll
    for (int r = 0; r < 8; r++) { Ssum[r] = 0.0f; mrow[r] = -__builtin_inff(); }

    _Float16* pb = sm + P_OFF + wid * 640;   // this wave's 16x(32+8) P buffer

    for (int et = 0; et < 16; et++) {
        const int e0 = et * 32;
        __syncthreads();

        // Stage 32 proj rows (row-major) via async global->LDS b128 copies.
        {
            const _Float16* gbase = projh + ((size_t)(b * TE_ + e0)) * H_;
#pragma unroll
            for (int i = 0; i < 4; i++) {
                int c = tid + 256 * i;       // 1024 chunks of 8 halfs
                int r = c >> 5, c8 = c & 31;
                async_ld_b128(lds_off(sm + PROJ_OFF + r * 264 + c8 * 8),
                              gbase, (unsigned)((r * H_ + c8 * 8) * 2));
            }
        }
        // Stage 32 encoder rows TRANSPOSED for context-GEMM B-fragments.
#pragma unroll
        for (int i = 0; i < 4; i++) {
            int c = tid + 256 * i;
            int r = c >> 5, c8 = c & 31;
            h8v v = *(const h8v*)(ench + ((size_t)(b * TE_ + e0 + r)) * H_ + c8 * 8);
#pragma unroll
            for (int j = 0; j < 8; j++) sm[ENCT_OFF + (c8 * 8 + j) * 40 + r] = v[j];
        }
        // Prefetch next e-tile (one 64B cacheline per thread covers 16 KB).
        if (et < 15) {
            const char* pp = (const char*)(projh + ((size_t)(b * TE_ + e0 + 32)) * H_);
            const char* pe = (const char*)(ench  + ((size_t)(b * TE_ + e0 + 32)) * H_);
            __builtin_prefetch(pp + tid * 64, 0, 1);
            __builtin_prefetch(pe + tid * 64, 0, 1);
        }
        // Fence our async LDS writes, then workgroup barrier.
        asm volatile("s_wait_asynccnt 0x0" ::: "memory");
        __syncthreads();

        // Scores: two 16x16 tiles over K=256.
        v8f s0 = zf, s1 = zf;
#pragma unroll
        for (int kc = 0; kc < 8; kc++) {
            v16h bf = load_bfrag(sm + PROJ_OFF + kc * 32, 264, lane);
            s0 = wmma_f16(afdec[kc], bf, s0);
        }
#pragma unroll
        for (int kc = 0; kc < 8; kc++) {
            v16h bf = load_bfrag(sm + PROJ_OFF + 16 * 264 + kc * 32, 264, lane);
            s1 = wmma_f16(afdec[kc], bf, s1);
        }

        // Running row-max (reduce over 16 lanes of each half-group).
        float smax[8];
#pragma unroll
        for (int r = 0; r < 8; r++) smax[r] = fmaxf(s0[r], s1[r]);
#pragma unroll
        for (int m = 1; m < 16; m <<= 1)
#pragma unroll
            for (int r = 0; r < 8; r++) smax[r] = fmaxf(smax[r], __shfl_xor(smax[r], m, 32));
        float scale[8];
#pragma unroll
        for (int r = 0; r < 8; r++) {
            float mn = fmaxf(mrow[r], smax[r]);
            scale[r] = __expf(mrow[r] - mn);
            mrow[r]  = mn;
        }
#pragma unroll
        for (int g = 0; g < 16; g++)
#pragma unroll
            for (int r = 0; r < 8; r++) Cacc[g][r] *= scale[r];

        // P = exp(s - m) * mask  (fits f16); accumulate row sums; stage to LDS.
        const float mk0 = mask[(size_t)b * TE_ + e0 + ln];
        const float mk1 = mask[(size_t)b * TE_ + e0 + 16 + ln];
#pragma unroll
        for (int r = 0; r < 8; r++) {
            float p0 = __expf(s0[r] - mrow[r]) * mk0;
            float p1 = __expf(s1[r] - mrow[r]) * mk1;
            Ssum[r] = Ssum[r] * scale[r] + p0 + p1;
            pb[(r + 8 * hi) * 40 + ln]      = (_Float16)p0;
            pb[(r + 8 * hi) * 40 + 16 + ln] = (_Float16)p1;
        }
        // Wave-private LDS RAW across lanes: wait on CDNA5 DS counter.
        asm volatile("s_wait_dscnt 0x0" ::: "memory");
        v16h pf = load_afrag(pb, 40, lane);   // A-fragment: 16 td x 32 e

        // Context accumulate: C_acc[16 x 256] += P @ enc_tile.
#pragma unroll
        for (int g = 0; g < 16; g++) {
            v16h bf = load_bfrag(sm + ENCT_OFF + g * 16 * 40, 40, lane);
            Cacc[g] = wmma_f16(pf, bf, Cacc[g]);
        }
    }

    // Row-sum reduction across the 16-lane group, then normalize & store.
#pragma unroll
    for (int m = 1; m < 16; m <<= 1)
#pragma unroll
        for (int r = 0; r < 8; r++) Ssum[r] += __shfl_xor(Ssum[r], m, 32);
    float rinv[8];
#pragma unroll
    for (int r = 0; r < 8; r++) rinv[r] = 1.0f / Ssum[r];

    const size_t rowbase = (size_t)(b * TD_ + td0 + wid * 16);
#pragma unroll
    for (int g = 0; g < 16; g++)
#pragma unroll
        for (int r = 0; r < 8; r++) {
            size_t row = rowbase + r + 8 * hi;
            out[row * (2 * H_) + g * 16 + ln] = Cacc[g][r] * rinv[r];
        }
}

// ---------------------------------------------------------------------------
extern "C" void kernel_launch(void* const* d_in, const int* in_sizes, int n_in,
                              void* d_out, int out_size, void* d_ws, size_t ws_size,
                              hipStream_t stream) {
    const float* enc  = (const float*)d_in[0];
    const float* dec  = (const float*)d_in[1];
    const float* mask = (const float*)d_in[2];
    const float* Wa   = (const float*)d_in[3];
    float* out = (float*)d_out;

    _Float16* ench  = (_Float16*)d_ws;                       // B*TE*H halfs (16 MB)
    _Float16* dech  = ench + (size_t)B_ * TE_ * H_;          // 16 MB
    _Float16* projh = dech + (size_t)B_ * TD_ * H_;          // 16 MB

    attn_convert_kernel<<<1024, 256, 0, stream>>>(enc, dec, ench, dech, out);

    const int projSmem = 128 * 264 * (int)sizeof(_Float16);  // 67584 B
    attn_proj_kernel<<<(B_ * TE_) / 128, 256, projSmem, stream>>>(ench, Wa, projh);

    const int attnSmem = (32 * 264 + 256 * 40 + 8 * 16 * 40) * (int)sizeof(_Float16); // 47616 B
    attn_fused_kernel<<<B_ * (TD_ / 128), 256, attnSmem, stream>>>(ench, dech, projh, mask, out);
}
